// GraphClsGraphSAGE_52621939310631
// MI455X (gfx1250) — matrix operand order repaired and checked by reading, not compiled, not run on hardware
//
#include <hip/hip_runtime.h>
#include <hip/hip_bf16.h>

typedef __attribute__((ext_vector_type(16))) __bf16 v16bf;
typedef __attribute__((ext_vector_type(8)))  float  v8f;

#define D_FEAT 128
#define N_CLS  10

// HW fp32 global atomic add (no return -> STOREcnt path), CDNA5 inline asm.
__device__ inline void atomic_add_f32(float* p, float v) {
  asm volatile("global_atomic_add_f32 %0, %1, off" :: "v"(p), "v"(v) : "memory");
}

// Async global->LDS copy, 16B per lane per issue (GLOBAL_LOAD_ASYNC_TO_LDS_B128,
// tracked by ASYNCcnt).  lds_addr is the LDS byte address (low 32 bits of the
// shared-aperture generic address).
__device__ inline void async_copy_b128(unsigned lds_addr, const void* gaddr) {
  asm volatile("global_load_async_to_lds_b128 %0, %1, off"
               :: "v"(lds_addr), "v"(gaddr) : "memory");
}

__device__ inline void wait_async0() {
#if __has_builtin(__builtin_amdgcn_s_wait_asynccnt)
  __builtin_amdgcn_s_wait_asynccnt(0);
#else
  asm volatile("s_wait_asynccnt 0x0" ::: "memory");
#endif
}

__global__ void fill_zero_kernel(float* __restrict__ p, int n) {
  int i = blockIdx.x * blockDim.x + threadIdx.x;
  if (i < n) p[i] = 0.0f;
}

__global__ void f2bf_kernel(const float* __restrict__ in, __bf16* __restrict__ out, int n) {
  int i = blockIdx.x * blockDim.x + threadIdx.x;
  if (i < n) out[i] = (__bf16)in[i];
}

__global__ void degree_kernel(const int* __restrict__ dst, float* __restrict__ deg, int E) {
  int i = blockIdx.x * blockDim.x + threadIdx.x;
  if (i < E) atomic_add_f32(&deg[dst[i]], 1.0f);
}

// One wave32 per edge: lane l moves h[src][4l..4l+3] into agg[dst][...] atomically.
__global__ void aggregate_kernel(const float* __restrict__ h, const int* __restrict__ src,
                                 const int* __restrict__ dst, float* __restrict__ agg, int E) {
  int wave = (int)((blockIdx.x * blockDim.x + threadIdx.x) >> 5);
  int lane = threadIdx.x & 31;
  if (wave >= E) return;
  int s = src[wave];
  int d = dst[wave];
  const float4 v = *(const float4*)(h + (size_t)s * D_FEAT + lane * 4);
  float* p = agg + (size_t)d * D_FEAT + lane * 4;
  atomic_add_f32(p + 0, v.x);
  atomic_add_f32(p + 1, v.y);
  atomic_add_f32(p + 2, v.z);
  atomic_add_f32(p + 3, v.w);
}

// A-fragment (16x32 bf16, MxK): lane<16 holds row M=lane, K = {0..7,16..23}+base;
// lane>=16 holds same row, K = {8..15,24..31}+base.  p already includes the
// row base and the +8 half-lane offset; pattern is p[0..7] and p[16..23].
__device__ inline v16bf load_a_frag(const float* __restrict__ p, float scale) {
  float4 x0 = *(const float4*)(p);
  float4 x1 = *(const float4*)(p + 4);
  float4 x2 = *(const float4*)(p + 16);
  float4 x3 = *(const float4*)(p + 20);
  v16bf a;
  a[0]  = (__bf16)(x0.x * scale); a[1]  = (__bf16)(x0.y * scale);
  a[2]  = (__bf16)(x0.z * scale); a[3]  = (__bf16)(x0.w * scale);
  a[4]  = (__bf16)(x1.x * scale); a[5]  = (__bf16)(x1.y * scale);
  a[6]  = (__bf16)(x1.z * scale); a[7]  = (__bf16)(x1.w * scale);
  a[8]  = (__bf16)(x2.x * scale); a[9]  = (__bf16)(x2.y * scale);
  a[10] = (__bf16)(x2.z * scale); a[11] = (__bf16)(x2.w * scale);
  a[12] = (__bf16)(x3.x * scale); a[13] = (__bf16)(x3.y * scale);
  a[14] = (__bf16)(x3.z * scale); a[15] = (__bf16)(x3.w * scale);
  return a;
}

// B-fragment (32x16 bf16, KxN) read from LDS: lane<16 = column N=lane,
// K=ks*32+{0..7,16..23}; lane>=16 = column N=lane-16, K=ks*32+{8..15,24..31}.
// Column o of B is row o of W (row-major [128][128]).
__device__ inline v16bf load_b_frag(const __bf16* __restrict__ W, int ot, int ks, int lane) {
  int o = ot * 16 + (lane & 15);
  int k = ks * 32 + ((lane & 16) ? 8 : 0);
  const __bf16* p = W + (size_t)o * D_FEAT + k;
  union { float4 f4[2]; v16bf v; } u;
  u.f4[0] = *(const float4*)(p);
  u.f4[1] = *(const float4*)(p + 16);
  return u.v;
}

// out[n,o] = relu( h[n,:]·Wself[o,:] + bself[o] + (agg[n,:]/max(deg,1))·Wneigh[o,:] )
// One wave per 16-node tile; weights staged per-block into LDS via async copies;
// 8 output col-tiles x (4+4) K-steps = 64 WMMAs/wave.
__global__ void __launch_bounds__(128)
sage_gemm_kernel(const float* __restrict__ h, const float* __restrict__ agg,
                 const float* __restrict__ deg,
                 const __bf16* __restrict__ Wself, const float* __restrict__ bself,
                 const __bf16* __restrict__ Wneigh,
                 float* __restrict__ out, int N) {
  __shared__ alignas(16) __bf16 sWs[D_FEAT * D_FEAT];   // 32 KB
  __shared__ alignas(16) __bf16 sWn[D_FEAT * D_FEAT];   // 32 KB

  // Stage both weight matrices: 2048 16B-chunks each, 128 threads -> 16 per thread.
  unsigned lbs = (unsigned)(uintptr_t)&sWs[0];
  unsigned lbn = (unsigned)(uintptr_t)&sWn[0];
#pragma unroll
  for (int i = 0; i < 16; ++i) {
    int c = (int)threadIdx.x + i * 128;                  // chunk id, 0..2047
    async_copy_b128(lbs + c * 16, (const char*)Wself + c * 16);
    async_copy_b128(lbn + c * 16, (const char*)Wneigh + c * 16);
  }
  wait_async0();
  __syncthreads();

  int wave = (int)((blockIdx.x * blockDim.x + threadIdx.x) >> 5);
  int lane = threadIdx.x & 31;
  int ntiles = (N + 15) >> 4;
  if (wave >= ntiles) return;   // after the barrier: no divergence hazard

  int rowbase = wave * 16;
  int laneM = lane & 15;
  int hi = lane >> 4;           // which K half-group this lane carries

  int nrow = rowbase + laneM;
  int nload = nrow < N ? nrow : N - 1;
  float dg = deg[nload];
  float scale = __builtin_amdgcn_rcpf(dg > 1.0f ? dg : 1.0f);  // 1/max(deg,1), HW rcp

  const float* hrow = h   + (size_t)nload * D_FEAT + hi * 8;
  const float* grow = agg + (size_t)nload * D_FEAT + hi * 8;

  v16bf aS[4], aN[4];
#pragma unroll
  for (int ks = 0; ks < 4; ++ks) {
    aS[ks] = load_a_frag(hrow + ks * 32, 1.0f);
    aN[ks] = load_a_frag(grow + ks * 32, scale);
  }

#pragma unroll
  for (int ot = 0; ot < 8; ++ot) {
    v8f acc = {0.f, 0.f, 0.f, 0.f, 0.f, 0.f, 0.f, 0.f};
#pragma unroll
    for (int ks = 0; ks < 4; ++ks) {
      v16bf b = load_b_frag(sWs, ot, ks, lane);
      acc = __builtin_amdgcn_wmma_f32_16x16x32_bf16(false, aS[ks], false, b,
                                                    (short)0, acc, false, false);
    }
#pragma unroll
    for (int ks = 0; ks < 4; ++ks) {
      v16bf b = load_b_frag(sWn, ot, ks, lane);
      acc = __builtin_amdgcn_wmma_f32_16x16x32_bf16(false, aN[ks], false, b,
                                                    (short)0, acc, false, false);
    }
    // C/D layout: VGPR j holds M = j + 8*hi, N = lane&15.
    int col = ot * 16 + laneM;
    float bias = bself[col];
#pragma unroll
    for (int j = 0; j < 8; ++j) {
      int row = rowbase + j + hi * 8;
      if (row < N) {
        float v = acc[j] + bias;
        out[(size_t)row * D_FEAT + col] = v > 0.0f ? v : 0.0f;
      }
    }
  }
}

// One wave per node; 10 dot-products of length 128, wave32 xor-reduction.
__global__ void classifier_kernel(const float* __restrict__ h, const float* __restrict__ Wcls,
                                  const float* __restrict__ bcls, float* __restrict__ out, int N) {
  int wave = (int)((blockIdx.x * blockDim.x + threadIdx.x) >> 5);
  int lane = threadIdx.x & 31;
  if (wave >= N) return;
  float4 hv = *(const float4*)(h + (size_t)wave * D_FEAT + lane * 4);
#pragma unroll
  for (int c = 0; c < N_CLS; ++c) {
    float4 wv = *(const float4*)(Wcls + (size_t)c * D_FEAT + lane * 4);
    float part = hv.x * wv.x + hv.y * wv.y + hv.z * wv.z + hv.w * wv.w;
#pragma unroll
    for (int off = 16; off > 0; off >>= 1) part += __shfl_xor(part, off, 32);
    if (lane == 0) out[(size_t)wave * N_CLS + c] = part + bcls[c];
  }
}

extern "C" void kernel_launch(void* const* d_in, const int* in_sizes, int n_in,
                              void* d_out, int out_size, void* d_ws, size_t ws_size,
                              hipStream_t stream) {
  const float* feat    = (const float*)d_in[0];
  const int*   src     = (const int*)d_in[1];
  const int*   dst     = (const int*)d_in[2];
  const float* Wself0  = (const float*)d_in[3];
  const float* bself0  = (const float*)d_in[4];
  const float* Wneigh0 = (const float*)d_in[5];
  const float* Wself1  = (const float*)d_in[6];
  const float* bself1  = (const float*)d_in[7];
  const float* Wneigh1 = (const float*)d_in[8];
  const float* Wcls    = (const float*)d_in[9];
  const float* bcls    = (const float*)d_in[10];
  float* out = (float*)d_out;

  const int N = in_sizes[0] / D_FEAT;   // T==1, so feat is [N,128]
  const int E = in_sizes[1];

  // Workspace carve-out (256B aligned): deg | agg | h1 | h2 | 4x bf16 weights.
  char* ws = (char*)d_ws;
  size_t cur = 0;
  auto carve = [&](size_t bytes) -> void* {
    void* p = ws + cur;
    cur = (cur + bytes + 255) & ~(size_t)255;
    return p;
  };
  float*  deg = (float*)carve((size_t)N * 4);
  float*  agg = (float*)carve((size_t)N * D_FEAT * 4);
  float*  h1  = (float*)carve((size_t)N * D_FEAT * 4);
  float*  h2  = (float*)carve((size_t)N * D_FEAT * 4);
  __bf16* wb0s = (__bf16*)carve((size_t)D_FEAT * D_FEAT * 2);
  __bf16* wb0n = (__bf16*)carve((size_t)D_FEAT * D_FEAT * 2);
  __bf16* wb1s = (__bf16*)carve((size_t)D_FEAT * D_FEAT * 2);
  __bf16* wb1n = (__bf16*)carve((size_t)D_FEAT * D_FEAT * 2);

  const int BT = 256;
  const int WN = D_FEAT * D_FEAT;
  const int aggN = N * D_FEAT;
  const int gemmBlocks = (((N + 15) / 16) + 3) / 4;     // 4 waves (tiles) per block
  const int edgeWaveBlocks = (int)(((size_t)E * 32 + BT - 1) / BT);

  // Degrees (shared by both layers).
  fill_zero_kernel<<<(N + BT - 1) / BT, BT, 0, stream>>>(deg, N);
  degree_kernel<<<(E + BT - 1) / BT, BT, 0, stream>>>(dst, deg, E);

  // Weights -> bf16 once.
  f2bf_kernel<<<(WN + BT - 1) / BT, BT, 0, stream>>>(Wself0,  wb0s, WN);
  f2bf_kernel<<<(WN + BT - 1) / BT, BT, 0, stream>>>(Wneigh0, wb0n, WN);
  f2bf_kernel<<<(WN + BT - 1) / BT, BT, 0, stream>>>(Wself1,  wb1s, WN);
  f2bf_kernel<<<(WN + BT - 1) / BT, BT, 0, stream>>>(Wneigh1, wb1n, WN);

  // Layer 0
  fill_zero_kernel<<<(aggN + BT - 1) / BT, BT, 0, stream>>>(agg, aggN);
  aggregate_kernel<<<edgeWaveBlocks, BT, 0, stream>>>(feat, src, dst, agg, E);
  sage_gemm_kernel<<<gemmBlocks, 128, 0, stream>>>(feat, agg, deg, wb0s, bself0, wb0n, h1, N);

  // Layer 1
  fill_zero_kernel<<<(aggN + BT - 1) / BT, BT, 0, stream>>>(agg, aggN);
  aggregate_kernel<<<edgeWaveBlocks, BT, 0, stream>>>(h1, src, dst, agg, E);
  sage_gemm_kernel<<<gemmBlocks, 128, 0, stream>>>(h1, agg, deg, wb1s, bself1, wb1n, h2, N);

  // Head
  classifier_kernel<<<(int)(((size_t)N * 32 + BT - 1) / BT), BT, 0, stream>>>(h2, Wcls, bcls, out, N);
}
